// Attention_32744830665235
// MI455X (gfx1250) — matrix operand order repaired and checked by reading, not compiled, not run on hardware
//
#include <hip/hip_runtime.h>
#include <hip/hip_bf16.h>

#define B_  4
#define S_  2048
#define D_  768
#define H_  12
#define HD_ 64
#define M_  (B_ * S_)   // 8192 rows

typedef __attribute__((ext_vector_type(16))) _Float16 v16h;
typedef __attribute__((ext_vector_type(8)))  _Float16 v8h;
typedef __attribute__((ext_vector_type(4)))  _Float16 v4h;
typedef __attribute__((ext_vector_type(8)))  float    v8f;

union Frag16 { v16h v; v8h h[2]; };

// ---------------------------------------------------------------------------
// Fragment loaders per CDNA5 ISA §7.12.2 (wave32, 16-bit data).
// A (16x32 MxK): lane holds row M=lane%16; element j(0..15):
//   K = (j/8)*16 + (lane/16)*8 + j%8  -> two contiguous 8-half (16B) chunks.
__device__ __forceinline__ v16h load_a_frag(const _Float16* row_ptr, int k0, int lane) {
  Frag16 f;
  const int off = k0 + ((lane >> 4) << 3);
  f.h[0] = *(const v8h*)(row_ptr + off);
  f.h[1] = *(const v8h*)(row_ptr + off + 16);
  return f.v;
}
// B (32x16 KxN): lane holds column N=lane%16; element j(0..15):
//   K = (j/8)*16 + (lane/16)*8 + j%8  -> two contiguous 8-half (16B) chunks.
// col_ptr points at the K-contiguous row for this lane's N.
__device__ __forceinline__ v16h load_b_frag(const _Float16* col_ptr, int k0, int lane) {
  Frag16 f;
  const int off = k0 + ((lane >> 4) << 3);
  f.h[0] = *(const v8h*)(col_ptr + off);
  f.h[1] = *(const v8h*)(col_ptr + off + 16);
  return f.v;
}
// C/D (16x16 f32): VGPR r, lane L -> M = r + 8*(L/16), N = L%16.

#define WMMA_F16(A, Bv, C) \
  __builtin_amdgcn_wmma_f32_16x16x32_f16(false, (A), false, (Bv), (short)0, (C), false, false)

// ---------------------------------------------------------------------------
__global__ void cvt_f32_to_f16_v4(const float* __restrict__ src,
                                  _Float16* __restrict__ dst, int n4) {
  const int i = blockIdx.x * blockDim.x + threadIdx.x;
  if (i < n4) {
    const float4 f = ((const float4*)src)[i];
    v4h h;
    h.x = (_Float16)f.x; h.y = (_Float16)f.y;
    h.z = (_Float16)f.z; h.w = (_Float16)f.w;
    ((v4h*)dst)[i] = h;
  }
}

// ---------------------------------------------------------------------------
// QKV projection: C[M,N] = X[M,K] * W[N,K]^T + bias. 4 waves/block, each wave
// computes 32x32; block tile 64x64. z selects Q/K/V. Q,K stored [B,H,S,HD];
// V stored transposed [B,H,HD,S] for contiguous PV B-fragments.
__global__ __launch_bounds__(128) void qkv_proj_kernel(
    const _Float16* __restrict__ x16,
    const _Float16* __restrict__ wq16, const _Float16* __restrict__ wk16,
    const _Float16* __restrict__ wv16,
    const float* __restrict__ qb, const float* __restrict__ kbs,
    const float* __restrict__ vb,
    _Float16* __restrict__ q16, _Float16* __restrict__ k16,
    _Float16* __restrict__ vT16)
{
  const int tid = threadIdx.x, lane = tid & 31, wave = tid >> 5;
  const int which = blockIdx.z;
  const _Float16* w   = (which == 0) ? wq16 : (which == 1) ? wk16 : wv16;
  const float*   bias = (which == 0) ? qb   : (which == 1) ? kbs  : vb;
  const int m0 = blockIdx.x * 64 + (wave >> 1) * 32;
  const int n0 = blockIdx.y * 64 + (wave & 1) * 32;

  const _Float16* arow0 = x16 + (size_t)(m0 + (lane & 15)) * D_;
  const _Float16* arow1 = arow0 + (size_t)16 * D_;
  const _Float16* brow0 = w + (size_t)(n0 + (lane & 15)) * D_;
  const _Float16* brow1 = brow0 + (size_t)16 * D_;

  v8f acc[2][2] = {};
  for (int k0 = 0; k0 < D_; k0 += 32) {
    const v16h a0 = load_a_frag(arow0, k0, lane);
    const v16h a1 = load_a_frag(arow1, k0, lane);
    const v16h b0 = load_b_frag(brow0, k0, lane);
    const v16h b1 = load_b_frag(brow1, k0, lane);
    acc[0][0] = WMMA_F16(a0, b0, acc[0][0]);
    acc[0][1] = WMMA_F16(a0, b1, acc[0][1]);
    acc[1][0] = WMMA_F16(a1, b0, acc[1][0]);
    acc[1][1] = WMMA_F16(a1, b1, acc[1][1]);
  }

#pragma unroll
  for (int i = 0; i < 2; ++i)
#pragma unroll
    for (int j = 0; j < 2; ++j) {
      const int n  = n0 + j * 16 + (lane & 15);
      const float bn = bias[n];
      const int h = n >> 6, hd = n & 63;
#pragma unroll
      for (int r = 0; r < 8; ++r) {
        const int m = m0 + i * 16 + r + ((lane >> 4) << 3);
        const int b = m >> 11, s = m & 2047;   // S_ == 2048
        const _Float16 hv = (_Float16)(acc[i][j][r] + bn);
        if (which == 2)
          vT16[(((size_t)b * H_ + h) * HD_ + hd) * S_ + s] = hv;
        else if (which == 0)
          q16[(((size_t)b * H_ + h) * S_ + s) * HD_ + hd] = hv;
        else
          k16[(((size_t)b * H_ + h) * S_ + s) * HD_ + hd] = hv;
      }
    }
}

// ---------------------------------------------------------------------------
// Flash attention: 4 waves/block, each wave owns 32 query rows of one (b,h).
// Online softmax over key tiles of 32. P re-swizzled D->A layout via LDS.
__global__ __launch_bounds__(128) void attn_kernel(
    const _Float16* __restrict__ q16, const _Float16* __restrict__ k16,
    const _Float16* __restrict__ vT16, const unsigned char* __restrict__ mask,
    _Float16* __restrict__ ctx16)
{
  __shared__ __attribute__((aligned(16))) _Float16 pbuf[4][32][32];
  const int tid = threadIdx.x, lane = tid & 31, wave = tid >> 5;
  const int b = blockIdx.z, h = blockIdx.y;
  const int q0 = blockIdx.x * 128 + wave * 32;

  const _Float16* qh = q16  + (size_t)(b * H_ + h) * S_ * HD_;
  const _Float16* kh = k16  + (size_t)(b * H_ + h) * S_ * HD_;
  const _Float16* vh = vT16 + (size_t)(b * H_ + h) * HD_ * S_;
  const unsigned char* mh = mask + (size_t)h * S_ * S_;   // mask is [1,H,S,S]

  // Q fragments: rows q0..q0+31, K-dim = HD = 64 -> 2 k-steps
  v16h qa[2][2];
#pragma unroll
  for (int i = 0; i < 2; ++i) {
    const _Float16* qrow = qh + (size_t)(q0 + i * 16 + (lane & 15)) * HD_;
#pragma unroll
    for (int kk = 0; kk < 2; ++kk) qa[i][kk] = load_a_frag(qrow, kk * 32, lane);
  }

  v8f o[2][4] = {};
  float mrow[2][8], lrow[2][8];
#pragma unroll
  for (int i = 0; i < 2; ++i)
#pragma unroll
    for (int r = 0; r < 8; ++r) { mrow[i][r] = -1e30f; lrow[i][r] = 0.f; }

  const float scale = 0.125f;   // 1/sqrt(64)

  for (int kt = 0; kt < S_; kt += 32) {
    // K fragments (B matrix of QK^T): "column" N = key position
    v16h kbf[2][2];
#pragma unroll
    for (int j = 0; j < 2; ++j) {
      const _Float16* krow = kh + (size_t)(kt + j * 16 + (lane & 15)) * HD_;
#pragma unroll
      for (int kk = 0; kk < 2; ++kk) kbf[j][kk] = load_b_frag(krow, kk * 32, lane);
    }
    v8f sc[2][2] = {};
#pragma unroll
    for (int i = 0; i < 2; ++i)
#pragma unroll
      for (int j = 0; j < 2; ++j) {
        sc[i][j] = WMMA_F16(qa[i][0], kbf[j][0], sc[i][j]);
        sc[i][j] = WMMA_F16(qa[i][1], kbf[j][1], sc[i][j]);
      }

    // mask + scale (D layout: row = r + 8*(lane/16), col = lane%16)
    float f[2][2][8];
#pragma unroll
    for (int i = 0; i < 2; ++i)
#pragma unroll
      for (int j = 0; j < 2; ++j) {
        const int n = kt + j * 16 + (lane & 15);
#pragma unroll
        for (int r = 0; r < 8; ++r) {
          const int m = q0 + i * 16 + r + ((lane >> 4) << 3);
          const unsigned char mk = mh[(size_t)m * S_ + n];
          f[i][j][r] = mk ? sc[i][j][r] * scale : -1e9f;
        }
      }

    // online softmax: row reductions across the 16-lane half-wave
#pragma unroll
    for (int i = 0; i < 2; ++i)
#pragma unroll
      for (int r = 0; r < 8; ++r) {
        float t = fmaxf(f[i][0][r], f[i][1][r]);
#pragma unroll
        for (int s = 1; s < 16; s <<= 1) t = fmaxf(t, __shfl_xor(t, s, 32));
        const float mn    = fmaxf(mrow[i][r], t);
        const float alpha = __expf(mrow[i][r] - mn);
        mrow[i][r] = mn;
        f[i][0][r] = __expf(f[i][0][r] - mn);
        f[i][1][r] = __expf(f[i][1][r] - mn);
        float rs = f[i][0][r] + f[i][1][r];
#pragma unroll
        for (int s = 1; s < 16; s <<= 1) rs += __shfl_xor(rs, s, 32);
        lrow[i][r] = lrow[i][r] * alpha + rs;
#pragma unroll
        for (int t4 = 0; t4 < 4; ++t4) o[i][t4][r] *= alpha;
      }

    // P (f32, D layout) -> LDS f16 row-major 32x32, then reload in A layout
#pragma unroll
    for (int i = 0; i < 2; ++i)
#pragma unroll
      for (int j = 0; j < 2; ++j)
#pragma unroll
        for (int r = 0; r < 8; ++r)
          pbuf[wave][i * 16 + r + ((lane >> 4) << 3)][j * 16 + (lane & 15)] =
              (_Float16)f[i][j][r];
    __syncthreads();
    v16h pa[2];
#pragma unroll
    for (int i = 0; i < 2; ++i)
      pa[i] = load_a_frag(&pbuf[wave][i * 16 + (lane & 15)][0], 0, lane);

    // V fragments (B matrix of P@V): "column" N = hd; V^T gives K-contiguity
    v16h vbf[4];
#pragma unroll
    for (int t4 = 0; t4 < 4; ++t4)
      vbf[t4] = load_b_frag(vh + (size_t)(t4 * 16 + (lane & 15)) * S_, kt, lane);
#pragma unroll
    for (int i = 0; i < 2; ++i)
#pragma unroll
      for (int t4 = 0; t4 < 4; ++t4)
        o[i][t4] = WMMA_F16(pa[i], vbf[t4], o[i][t4]);
    __syncthreads();
  }

  // normalize and emit ctx in [B,S,D] f16 (D index = h*64 + hd)
#pragma unroll
  for (int i = 0; i < 2; ++i)
#pragma unroll
    for (int t4 = 0; t4 < 4; ++t4) {
      const int hd = t4 * 16 + (lane & 15);
#pragma unroll
      for (int r = 0; r < 8; ++r) {
        const int s = q0 + i * 16 + r + ((lane >> 4) << 3);
        const float val = o[i][t4][r] / lrow[i][r];
        ctx16[(size_t)(b * S_ + s) * D_ + h * HD_ + hd] = (_Float16)val;
      }
    }
}

// ---------------------------------------------------------------------------
// Output projection: out[M,N] = ctx[M,K] * Wo[N,K]^T + bias, f32 store.
__global__ __launch_bounds__(128) void out_proj_kernel(
    const _Float16* __restrict__ ctx16, const _Float16* __restrict__ wo16,
    const float* __restrict__ ob, float* __restrict__ out)
{
  const int tid = threadIdx.x, lane = tid & 31, wave = tid >> 5;
  const int m0 = blockIdx.x * 64 + (wave >> 1) * 32;
  const int n0 = blockIdx.y * 64 + (wave & 1) * 32;

  const _Float16* arow0 = ctx16 + (size_t)(m0 + (lane & 15)) * D_;
  const _Float16* arow1 = arow0 + (size_t)16 * D_;
  const _Float16* brow0 = wo16 + (size_t)(n0 + (lane & 15)) * D_;
  const _Float16* brow1 = brow0 + (size_t)16 * D_;

  v8f acc[2][2] = {};
  for (int k0 = 0; k0 < D_; k0 += 32) {
    const v16h a0 = load_a_frag(arow0, k0, lane);
    const v16h a1 = load_a_frag(arow1, k0, lane);
    const v16h b0 = load_b_frag(brow0, k0, lane);
    const v16h b1 = load_b_frag(brow1, k0, lane);
    acc[0][0] = WMMA_F16(a0, b0, acc[0][0]);
    acc[0][1] = WMMA_F16(a0, b1, acc[0][1]);
    acc[1][0] = WMMA_F16(a1, b0, acc[1][0]);
    acc[1][1] = WMMA_F16(a1, b1, acc[1][1]);
  }
#pragma unroll
  for (int i = 0; i < 2; ++i)
#pragma unroll
    for (int j = 0; j < 2; ++j) {
      const int n = n0 + j * 16 + (lane & 15);
      const float bn = ob[n];
#pragma unroll
      for (int r = 0; r < 8; ++r) {
        const int m = m0 + i * 16 + r + ((lane >> 4) << 3);
        out[(size_t)m * D_ + n] = acc[i][j][r] + bn;
      }
    }
}

// ---------------------------------------------------------------------------
extern "C" void kernel_launch(void* const* d_in, const int* in_sizes, int n_in,
                              void* d_out, int out_size, void* d_ws, size_t ws_size,
                              hipStream_t stream) {
  (void)in_sizes; (void)n_in; (void)out_size; (void)ws_size;

  const float*         hidden = (const float*)d_in[0];
  const unsigned char* mask   = (const unsigned char*)d_in[1];   // bool -> u8
  const float* q_w = (const float*)d_in[2];
  const float* q_b = (const float*)d_in[3];
  const float* k_w = (const float*)d_in[4];
  const float* k_b = (const float*)d_in[5];
  const float* v_w = (const float*)d_in[6];
  const float* v_b = (const float*)d_in[7];
  const float* o_w = (const float*)d_in[8];
  const float* o_b = (const float*)d_in[9];
  float* out = (float*)d_out;

  // workspace bump allocator (256B aligned)
  char* ws = (char*)d_ws;
  auto alloc = [&](size_t bytes) -> void* {
    void* p = ws; ws += (bytes + 255) & ~(size_t)255; return p;
  };
  _Float16* x16  = (_Float16*)alloc((size_t)M_ * D_ * 2);
  _Float16* wq16 = (_Float16*)alloc((size_t)D_ * D_ * 2);
  _Float16* wk16 = (_Float16*)alloc((size_t)D_ * D_ * 2);
  _Float16* wv16 = (_Float16*)alloc((size_t)D_ * D_ * 2);
  _Float16* wo16 = (_Float16*)alloc((size_t)D_ * D_ * 2);
  _Float16* q16  = (_Float16*)alloc((size_t)M_ * D_ * 2);   // [B,H,S,HD]
  _Float16* k16  = (_Float16*)alloc((size_t)M_ * D_ * 2);   // [B,H,S,HD]
  _Float16* vT16 = (_Float16*)alloc((size_t)M_ * D_ * 2);   // [B,H,HD,S]
  _Float16* c16  = (_Float16*)alloc((size_t)M_ * D_ * 2);   // ctx [B,S,D]

  auto cvt = [&](const float* s, _Float16* d, int nelem) {
    const int n4 = nelem / 4;
    cvt_f32_to_f16_v4<<<(n4 + 255) / 256, 256, 0, stream>>>(s, d, n4);
  };
  cvt(hidden, x16, M_ * D_);
  cvt(q_w, wq16, D_ * D_);
  cvt(k_w, wk16, D_ * D_);
  cvt(v_w, wv16, D_ * D_);
  cvt(o_w, wo16, D_ * D_);

  qkv_proj_kernel<<<dim3(M_ / 64, D_ / 64, 3), 128, 0, stream>>>(
      x16, wq16, wk16, wv16, q_b, k_b, v_b, q16, k16, vT16);

  attn_kernel<<<dim3(S_ / 128, H_, B_), 128, 0, stream>>>(
      q16, k16, vT16, mask, c16);

  out_proj_kernel<<<dim3(M_ / 64, D_ / 64, 1), 128, 0, stream>>>(
      c16, wo16, o_b, out);
}
